// RTEncoderOnly_68195490726196
// MI455X (gfx1250) — compile-verified, matrix operand order checked by reading, not run on hardware
//
#include <hip/hip_runtime.h>
#include <stdint.h>
#include <stddef.h>

#define B_   2
#define S_   2048
#define D_   512
#define H_   8
#define L_   4
#define DFF_ 2048
#define DT_  384
#define DH_  64
#define MS_  (B_*S_)     // 4096 rows
#define SW_  (S_/32)     // 64 mask words per row

typedef __attribute__((ext_vector_type(16))) __bf16 v16bf;
typedef __attribute__((ext_vector_type(8)))  float  v8f;

struct Frag { uint32_t w[8]; };

__device__ inline v16bf frag_as_vec(const Frag& f) {
  union { Frag fr; v16bf v; } u; u.fr = f; return u.v;
}

__device__ inline uint16_t f2bf(float f) {
  uint32_t u = __builtin_bit_cast(uint32_t, f);
  uint32_t r = u + 0x7fffu + ((u >> 16) & 1u);
  return (uint16_t)(r >> 16);
}
__device__ inline uint32_t pack2(float a, float b) {
  return (uint32_t)f2bf(a) | ((uint32_t)f2bf(b) << 16);
}

__device__ inline v8f vzero8() { v8f z = {0.f,0.f,0.f,0.f,0.f,0.f,0.f,0.f}; return z; }

__device__ inline v8f wmma_bf16(const Frag& a, const Frag& b, v8f c) {
  return __builtin_amdgcn_wmma_f32_16x16x32_bf16(
      /*neg_a=*/false, frag_as_vec(a), /*neg_b=*/false, frag_as_vec(b),
      /*c_mod=*/(short)0, c, /*reuse_a=*/false, /*reuse_b=*/false);
}

// Load a 16x32 bf16 A-style fragment (also used for B-style, symmetric layout):
// lane L (0-15): row = `row`, K = k0 + {0..7} in w[0..3], k0 + {16..23} in w[4..7]
// lane L (16-31): same rows, K offset +8 / +24.
__device__ inline Frag load_frag(const uint16_t* __restrict__ base, int row, int ld,
                                 int k0, int lane) {
  const uint16_t* p = base + (size_t)row * ld + k0 + ((lane >> 4) << 3);
  uint4 lo = *(const uint4*)p;
  uint4 hi = *(const uint4*)(p + 16);
  Frag f;
  f.w[0]=lo.x; f.w[1]=lo.y; f.w[2]=lo.z; f.w[3]=lo.w;
  f.w[4]=hi.x; f.w[5]=hi.y; f.w[6]=hi.z; f.w[7]=hi.w;
  return f;
}

// Fragment load from a precomputed row pointer (row*ld folded in), lane-half
// K offset klo = (lane>>4)<<3 passed in.
__device__ inline Frag load_frag_p(const uint16_t* __restrict__ rowp, int k0, int klo) {
  const uint16_t* p = rowp + k0 + klo;
  uint4 lo = *(const uint4*)p;
  uint4 hi = *(const uint4*)(p + 16);
  Frag f;
  f.w[0]=lo.x; f.w[1]=lo.y; f.w[2]=lo.z; f.w[3]=lo.w;
  f.w[4]=hi.x; f.w[5]=hi.y; f.w[6]=hi.z; f.w[7]=hi.w;
  return f;
}

__device__ inline float gelu_tanh(float x) {
  float x3 = x * x * x;
  return 0.5f * x * (1.f + tanhf(0.7978845608028654f * (x + 0.044715f * x3)));
}

// ---------------------------------------------------------------- prep kernels
__global__ void cvt_bf16_kernel(const float* __restrict__ src, uint16_t* __restrict__ dst, int n) {
  int i = blockIdx.x * blockDim.x + threadIdx.x;
  if (i < n) dst[i] = f2bf(src[i]);
}

// W (K x N) f32  ->  WT (N x K) bf16
__global__ void transpose_bf16_kernel(const float* __restrict__ W, uint16_t* __restrict__ WT,
                                      int K, int N) {
  int i = blockIdx.x * blockDim.x + threadIdx.x;
  if (i >= K * N) return;
  int n = i / K, k = i % K;
  WT[i] = f2bf(W[(size_t)k * N + n]);
}

// ---------------------------------------------------------------- mask kernel
// packed masks: [b][rel(4)][q][S/32] uint32
__global__ void build_masks_kernel(const int* __restrict__ node, const int* __restrict__ f2p,
                                   const int* __restrict__ col, const int* __restrict__ tab,
                                   const uint8_t* __restrict__ pad, uint32_t* __restrict__ mw) {
  int idx = blockIdx.x * blockDim.x + threadIdx.x;
  if (idx >= B_ * S_ * SW_) return;
  int kw = idx % SW_;
  int q  = (idx / SW_) % S_;
  int b  = idx / (SW_ * S_);
  int rq = b * S_ + q;
  bool qp = (pad[rq] == 0);
  int nq = node[rq], cq = col[rq], tq = tab[rq];
  int fq[5];
#pragma unroll
  for (int i = 0; i < 5; ++i) fq[i] = f2p[(size_t)rq * 5 + i];
  uint32_t m0 = 0, m1 = 0, m2 = 0, m3 = 0;
  if (qp) {
    for (int j = 0; j < 32; ++j) {
      int k = kw * 32 + j;
      int rk = b * S_ + k;
      if (pad[rk]) continue;
      int nk = node[rk];
      bool same = (nk == nq);
      bool kvin = false, qin = false;
#pragma unroll
      for (int i = 0; i < 5; ++i) {
        kvin |= (nk == fq[i]);
        qin  |= (nq == f2p[(size_t)rk * 5 + i]);
      }
      bool ct = (col[rk] == cq) && (tab[rk] == tq);
      uint32_t bit = 1u << j;
      if (same || kvin) m0 |= bit;
      if (qin)          m1 |= bit;
      if (ct)           m2 |= bit;
      m3 |= bit;
    }
  }
  size_t base = (((size_t)b * 4) * S_ + q) * SW_ + kw;
  size_t rs = (size_t)S_ * SW_;
  mw[base] = m0; mw[base + rs] = m1; mw[base + 2*rs] = m2; mw[base + 3*rs] = m3;
}

// ---------------------------------------------------------------- GEMM
// C(MxN) = A(MxK,bf16 row-major) * BT(NxK,bf16 row-major)^T.
// Per-wave tile: 32(M) x 64(N): 2 A-frags x 4 B-frags -> 8 WMMA per 32-k step
// from 12 b128 loads. Single-buffered (6 live frags + 64 acc VGPRs, no spills);
// latency hidden by multi-wave occupancy.
// Epilogue modes:
//  0: outF[m*N+n] += acc                      (residual add, Wo)
//  2: outH[m*N+n]  = bf16(gelu(acc+bias[n]))  (FFN1)
//  3: QK scatter -> outH[((b*H+h)*S+s)*DH+dh]
//  4: Vt scatter -> outH[((b*H+h)*DH+dh)*S+s]
//  5: outF[m*N+n]  = acc                      (encoder temps)
//  7: outF[m*N+n] += acc + bias[n]            (FFN2 residual)
__global__ void gemm_bf16_kernel(const uint16_t* __restrict__ A, const uint16_t* __restrict__ BT,
                                 const float* __restrict__ bias,
                                 float* __restrict__ outF, uint16_t* __restrict__ outH,
                                 int M, int N, int K, int mode) {
  int lane = threadIdx.x;
  int m0 = (blockIdx.y * 4 + threadIdx.y) * 32;
  int n0 = blockIdx.x * 64;
  if (m0 >= M) return;
  int lq = lane & 15, hi = lane >> 4;
  int klo = hi << 3;

  const uint16_t* ar0 = A + (size_t)(m0 + lq) * K;
  const uint16_t* ar1 = A + (size_t)(m0 + 16 + lq) * K;
  const uint16_t* br0 = BT + (size_t)(n0 + lq) * K;
  const uint16_t* br1 = BT + (size_t)(n0 + 16 + lq) * K;
  const uint16_t* br2 = BT + (size_t)(n0 + 32 + lq) * K;
  const uint16_t* br3 = BT + (size_t)(n0 + 48 + lq) * K;

  v8f acc[2][4];
#pragma unroll
  for (int i = 0; i < 2; ++i)
#pragma unroll
    for (int t = 0; t < 4; ++t) acc[i][t] = vzero8();

  for (int k0 = 0; k0 < K; k0 += 32) {
    Frag a0 = load_frag_p(ar0, k0, klo);
    Frag a1 = load_frag_p(ar1, k0, klo);
    Frag b0 = load_frag_p(br0, k0, klo);
    Frag b1 = load_frag_p(br1, k0, klo);
    Frag b2 = load_frag_p(br2, k0, klo);
    Frag b3 = load_frag_p(br3, k0, klo);

    acc[0][0] = wmma_bf16(a0, b0, acc[0][0]);
    acc[1][0] = wmma_bf16(a1, b0, acc[1][0]);
    acc[0][1] = wmma_bf16(a0, b1, acc[0][1]);
    acc[1][1] = wmma_bf16(a1, b1, acc[1][1]);
    acc[0][2] = wmma_bf16(a0, b2, acc[0][2]);
    acc[1][2] = wmma_bf16(a1, b2, acc[1][2]);
    acc[0][3] = wmma_bf16(a0, b3, acc[0][3]);
    acc[1][3] = wmma_bf16(a1, b3, acc[1][3]);
  }

#pragma unroll
  for (int i = 0; i < 2; ++i) {
#pragma unroll
    for (int t = 0; t < 4; ++t) {
#pragma unroll
      for (int r = 0; r < 8; ++r) {
        int m = m0 + 16 * i + r + (hi << 3);
        int n = n0 + 16 * t + lq;
        float v = acc[i][t][r];
        if (mode == 0) {
          outF[(size_t)m * N + n] += v;
        } else if (mode == 2) {
          outH[(size_t)m * N + n] = f2bf(gelu_tanh(v + bias[n]));
        } else if (mode == 3) {
          int b = m >> 11, s = m & (S_ - 1);
          int h = n >> 6, dh = n & 63;
          outH[(((size_t)(b * H_ + h)) * S_ + s) * DH_ + dh] = f2bf(v);
        } else if (mode == 4) {
          int b = m >> 11, s = m & (S_ - 1);
          int h = n >> 6, dh = n & 63;
          outH[(((size_t)(b * H_ + h)) * DH_ + dh) * S_ + s] = f2bf(v);
        } else if (mode == 5) {
          outF[(size_t)m * N + n] = v;
        } else { // 7
          outF[(size_t)m * N + n] += v + bias[n];
        }
      }
    }
  }
}

// ---------------------------------------------------------------- LayerNorm
__device__ inline void block_stats_512(float v0, float v1, float& mean, float& rstd, float* sh) {
  float s = v0 + v1, s2 = v0 * v0 + v1 * v1;
#pragma unroll
  for (int off = 16; off; off >>= 1) {
    s  += __shfl_xor(s,  off, 32);
    s2 += __shfl_xor(s2, off, 32);
  }
  int wid = threadIdx.x >> 5;
  if ((threadIdx.x & 31) == 0) { sh[wid] = s; sh[8 + wid] = s2; }
  __syncthreads();
  float ts = 0.f, ts2 = 0.f;
#pragma unroll
  for (int w = 0; w < 8; ++w) { ts += sh[w]; ts2 += sh[8 + w]; }
  mean = ts * (1.f / 512.f);
  float var = ts2 * (1.f / 512.f) - mean * mean;
  rstd = rsqrtf(var + 1e-5f);
  __syncthreads();
}

// One block (256 thr) per row; writes bf16 (outH) or f32 (outF)
__global__ void ln_rows_kernel(const float* __restrict__ x, const float* __restrict__ g,
                               const float* __restrict__ bt, uint16_t* __restrict__ outH,
                               float* __restrict__ outF) {
  __shared__ float sh[16];
  int row = blockIdx.x, t = threadIdx.x;
  const float* xr = x + (size_t)row * D_;
  float v0 = xr[t], v1 = xr[t + 256];
  float mean, rstd;
  block_stats_512(v0, v1, mean, rstd, sh);
  float y0 = (v0 - mean) * rstd * g[t]       + bt[t];
  float y1 = (v1 - mean) * rstd * g[t + 256] + bt[t + 256];
  if (outH) {
    outH[(size_t)row * D_ + t] = f2bf(y0);
    outH[(size_t)row * D_ + t + 256] = f2bf(y1);
  } else {
    outF[(size_t)row * D_ + t] = y0;
    outF[(size_t)row * D_ + t + 256] = y1;
  }
}

// ---------------------------------------------------------------- encoder pointwise
__global__ void encode_pointwise_kernel(
    const float* __restrict__ tmp_col, const float* __restrict__ tmp_text,
    const float* __restrict__ b_col, const float* __restrict__ w_num, const float* __restrict__ b_num,
    const float* __restrict__ b_text, const float* __restrict__ w_dt, const float* __restrict__ b_dt,
    const float* __restrict__ w_bool, const float* __restrict__ b_bool,
    const float* __restrict__ ln_g, const float* __restrict__ ln_b, const float* __restrict__ memb,
    const float* __restrict__ numv, const float* __restrict__ dtv, const float* __restrict__ boolv,
    const int* __restrict__ sem, const uint8_t* __restrict__ msk, const uint8_t* __restrict__ pad,
    float* __restrict__ x) {
  __shared__ float sh[16];
  int m = blockIdx.x, t = threadIdx.x;
  bool np = (pad[m] == 0);
  float npf = np ? 1.f : 0.f;

  float c0 = tmp_col[(size_t)m * D_ + t]       + b_col[t];
  float c1 = tmp_col[(size_t)m * D_ + t + 256] + b_col[t + 256];
  float mean, rstd;
  block_stats_512(c0, c1, mean, rstd, sh);
  float x0 = ((c0 - mean) * rstd * ln_g[t]       + ln_b[t])       * npf;
  float x1 = ((c1 - mean) * rstd * ln_g[t + 256] + ln_b[t + 256]) * npf;

  int tt = sem[m];
  bool mk = (msk[m] != 0);
  if (np && !mk) {
    float v0, v1;
    if (tt == 0)      { float nv = numv[m];  v0 = nv * w_num[t]  + b_num[t];  v1 = nv * w_num[t+256]  + b_num[t+256]; }
    else if (tt == 1) { v0 = tmp_text[(size_t)m*D_ + t] + b_text[t]; v1 = tmp_text[(size_t)m*D_ + t + 256] + b_text[t+256]; }
    else if (tt == 2) { float dv = dtv[m];   v0 = dv * w_dt[t]   + b_dt[t];   v1 = dv * w_dt[t+256]   + b_dt[t+256]; }
    else              { float bv = boolv[m]; v0 = bv * w_bool[t] + b_bool[t]; v1 = bv * w_bool[t+256] + b_bool[t+256]; }
    block_stats_512(v0, v1, mean, rstd, sh);
    const float* g  = ln_g + (size_t)(tt + 1) * D_;
    const float* bb = ln_b + (size_t)(tt + 1) * D_;
    x0 += (v0 - mean) * rstd * g[t]       + bb[t];
    x1 += (v1 - mean) * rstd * g[t + 256] + bb[t + 256];
  } else if (np && mk) {
    x0 += memb[(size_t)tt * D_ + t];
    x1 += memb[(size_t)tt * D_ + t + 256];
  }
  x[(size_t)m * D_ + t] = x0;
  x[(size_t)m * D_ + t + 256] = x1;
}

// ---------------------------------------------------------------- attention
// Flash attention over S^T = K * Q^T so that exp(P) repacks into WMMA A-frag
// with no cross-lane movement. Q/K row-major (BH,S,DH) bf16; V transposed
// (BH,DH,S) bf16. Output bf16 (B,S,D).
__global__ void attention_kernel(const uint16_t* __restrict__ Q, const uint16_t* __restrict__ Kd,
                                 const uint16_t* __restrict__ Vt, const uint32_t* __restrict__ mw,
                                 uint16_t* __restrict__ Aout) {
  int lane = threadIdx.x;
  int lq = lane & 15, hi = lane >> 4;
  int bh = blockIdx.y;
  int b = bh / H_, h = bh % H_, rel = h >> 1;
  int q0 = (blockIdx.x * 4 + threadIdx.y) * 16;
  int q = q0 + lq;
  const float scale = 0.125f; // 1/sqrt(64)

  const uint16_t* Qb = Q  + (size_t)bh * S_ * DH_;
  const uint16_t* Kb = Kd + (size_t)bh * S_ * DH_;
  const uint16_t* Vb = Vt + (size_t)bh * DH_ * S_;
  const uint32_t* mwr = mw + (((size_t)(b * 4 + rel)) * S_ + q) * SW_;

  // Q as B-fragment of S^T = K * Q^T (lane n = query, contiguous dh)
  Frag bq0 = load_frag(Qb, q, DH_, 0, lane);
  Frag bq1 = load_frag(Qb, q, DH_, 32, lane);

  float m_run = -1e30f, l_run = 0.f;
  v8f acc[4];
#pragma unroll
  for (int t = 0; t < 4; ++t) acc[t] = vzero8();

  for (int kb = 0; kb < S_; kb += 32) {
    uint32_t w = mwr[kb >> 5];
    if (__ballot(w != 0) == 0ull) continue;

    // S^T tiles: st0 = kv rows kb..kb+15, st1 = kb+16..kb+31 (x 16 queries)
    v8f st0 = vzero8(), st1 = vzero8();
    {
      Frag a = load_frag(Kb, kb + lq, DH_, 0, lane);
      st0 = wmma_bf16(a, bq0, st0);
      a = load_frag(Kb, kb + lq, DH_, 32, lane);
      st0 = wmma_bf16(a, bq1, st0);
      a = load_frag(Kb, kb + 16 + lq, DH_, 0, lane);
      st1 = wmma_bf16(a, bq0, st1);
      a = load_frag(Kb, kb + 16 + lq, DH_, 32, lane);
      st1 = wmma_bf16(a, bq1, st1);
    }

    int off = hi << 3;
    float s0[8], s1[8];
    float mb = -1e30f;
#pragma unroll
    for (int r = 0; r < 8; ++r) {
      s0[r] = ((w >> (off + r)) & 1u)      ? st0[r] * scale : -1e30f;
      s1[r] = ((w >> (16 + off + r)) & 1u) ? st1[r] * scale : -1e30f;
      mb = fmaxf(mb, fmaxf(s0[r], s1[r]));
    }
    mb = fmaxf(mb, __shfl_xor(mb, 16, 32));
    float mnew = fmaxf(m_run, mb);
    float alpha = __expf(m_run - mnew);

    float p0[8], p1[8], ps = 0.f;
#pragma unroll
    for (int r = 0; r < 8; ++r) {
      p0[r] = (s0[r] > -1e29f) ? __expf(s0[r] - mnew) : 0.f;
      p1[r] = (s1[r] > -1e29f) ? __expf(s1[r] - mnew) : 0.f;
      ps += p0[r] + p1[r];
    }
    ps += __shfl_xor(ps, 16, 32);
    l_run = l_run * alpha + ps;
    m_run = mnew;

    // P -> A fragment (lane-half K split matches A layout exactly)
    Frag ap;
#pragma unroll
    for (int j = 0; j < 4; ++j) {
      ap.w[j]     = pack2(p0[2 * j], p0[2 * j + 1]);
      ap.w[4 + j] = pack2(p1[2 * j], p1[2 * j + 1]);
    }

    // rescale accumulators: per C-VGPR row r, q-row = q0 + r + 8*hi
    float ar[8];
#pragma unroll
    for (int r = 0; r < 8; ++r) ar[r] = __shfl(alpha, r + (hi << 3), 32);
#pragma unroll
    for (int t = 0; t < 4; ++t)
#pragma unroll
      for (int r = 0; r < 8; ++r) acc[t][r] *= ar[r];

    // O += P(16xq,32kv) * V(32kv,64dh)
#pragma unroll
    for (int t = 0; t < 4; ++t) {
      Frag bv = load_frag(Vb, 16 * t + lq, S_, kb, lane);
      acc[t] = wmma_bf16(ap, bv, acc[t]);
    }
  }

  float linv[8];
#pragma unroll
  for (int r = 0; r < 8; ++r) {
    float li = __shfl(l_run, r + (hi << 3), 32);
    linv[r] = (li > 0.f) ? 1.f / li : 0.f; // l==0 -> fully masked row -> zero out
  }
#pragma unroll
  for (int t = 0; t < 4; ++t) {
#pragma unroll
    for (int r = 0; r < 8; ++r) {
      int qq = q0 + r + (hi << 3);
      int dh = 16 * t + lq;
      Aout[((size_t)b * S_ + qq) * D_ + h * DH_ + dh] = f2bf(acc[t][r] * linv[r]);
    }
  }
}

// ---------------------------------------------------------------- launch
static inline int cdiv(int a, int b) { return (a + b - 1) / b; }

extern "C" void kernel_launch(void* const* d_in, const int* in_sizes, int n_in,
                              void* d_out, int out_size, void* d_ws, size_t ws_size,
                              hipStream_t stream) {
  (void)in_sizes; (void)n_in; (void)out_size; (void)ws_size;

  // inputs (setup_inputs dict order)
  const int*     node   = (const int*)d_in[0];
  const int*     f2p    = (const int*)d_in[1];
  const int*     col    = (const int*)d_in[2];
  const int*     tab    = (const int*)d_in[3];
  const int*     sem    = (const int*)d_in[4];
  const uint8_t* msk    = (const uint8_t*)d_in[5];
  const uint8_t* pad    = (const uint8_t*)d_in[6];
  const float*   numv   = (const float*)d_in[7];
  const float*   textv  = (const float*)d_in[8];
  const float*   dtv    = (const float*)d_in[9];
  const float*   boolv  = (const float*)d_in[10];
  const float*   colv   = (const float*)d_in[11];
  const float*   w_col  = (const float*)d_in[12];
  const float*   b_col  = (const float*)d_in[13];
  const float*   w_num  = (const float*)d_in[14];
  const float*   b_num  = (const float*)d_in[15];
  const float*   w_text = (const float*)d_in[16];
  const float*   b_text = (const float*)d_in[17];
  const float*   w_dt   = (const float*)d_in[18];
  const float*   b_dt   = (const float*)d_in[19];
  const float*   w_bool = (const float*)d_in[20];
  const float*   b_bool = (const float*)d_in[21];
  const float*   ln_g   = (const float*)d_in[22];
  const float*   ln_b   = (const float*)d_in[23];
  const float*   memb   = (const float*)d_in[24];
  const float*   Wq     = (const float*)d_in[25];
  const float*   Wk     = (const float*)d_in[26];
  const float*   Wv     = (const float*)d_in[27];
  const float*   Wo     = (const float*)d_in[28];
  const float*   aln_g  = (const float*)d_in[29];
  const float*   aln_b  = (const float*)d_in[30];
  const float*   W1     = (const float*)d_in[31];
  const float*   b1     = (const float*)d_in[32];
  const float*   W2     = (const float*)d_in[33];
  const float*   b2     = (const float*)d_in[34];
  const float*   fln_g  = (const float*)d_in[35];
  const float*   fln_b  = (const float*)d_in[36];
  const float*   out_g  = (const float*)d_in[37];
  const float*   out_b  = (const float*)d_in[38];

  // workspace carve-up
  char* ws = (char*)d_ws;
  size_t off = 0;
  auto alloc = [&](size_t bytes) { size_t o = off; off += (bytes + 255) & ~(size_t)255; return o; };
  uint32_t* maskw  = (uint32_t*)(ws + alloc((size_t)B_*4*S_*SW_*4));
  float*    x      = (float*)   (ws + alloc((size_t)MS_*D_*4));
  uint16_t* hbuf   = (uint16_t*)(ws + alloc((size_t)MS_*D_*2));
  uint16_t* colbf  = (uint16_t*)(ws + alloc((size_t)MS_*DT_*2));
  uint16_t* textbf = (uint16_t*)(ws + alloc((size_t)MS_*DT_*2));
  float*    tcol   = (float*)   (ws + alloc((size_t)MS_*D_*4));
  float*    ttext  = (float*)   (ws + alloc((size_t)MS_*D_*4));
  uint16_t* wcolT  = (uint16_t*)(ws + alloc((size_t)D_*DT_*2));
  uint16_t* wtextT = (uint16_t*)(ws + alloc((size_t)D_*DT_*2));
  uint16_t* WqT    = (uint16_t*)(ws + alloc((size_t)L_*D_*D_*2));
  uint16_t* WkT    = (uint16_t*)(ws + alloc((size_t)L_*D_*D_*2));
  uint16_t* WvT    = (uint16_t*)(ws + alloc((size_t)L_*D_*D_*2));
  uint16_t* WoT    = (uint16_t*)(ws + alloc((size_t)L_*D_*D_*2));
  uint16_t* W1T    = (uint16_t*)(ws + alloc((size_t)L_*DFF_*D_*2));
  uint16_t* W2T    = (uint16_t*)(ws + alloc((size_t)L_*D_*DFF_*2));
  uint16_t* Qb     = (uint16_t*)(ws + alloc((size_t)B_*H_*S_*DH_*2));
  uint16_t* Kb     = (uint16_t*)(ws + alloc((size_t)B_*H_*S_*DH_*2));
  uint16_t* Vtb    = (uint16_t*)(ws + alloc((size_t)B_*H_*DH_*S_*2));
  uint16_t* attnb  = (uint16_t*)(ws + alloc((size_t)MS_*D_*2));
  uint16_t* ffnb   = (uint16_t*)(ws + alloc((size_t)MS_*DFF_*2));

  // ---- weight prep (bf16, transposed to N x K) ----
  {
    int n = DT_ * D_;
    transpose_bf16_kernel<<<cdiv(n,256),256,0,stream>>>(w_col,  wcolT,  DT_, D_);
    transpose_bf16_kernel<<<cdiv(n,256),256,0,stream>>>(w_text, wtextT, DT_, D_);
  }
  for (int l = 0; l < L_; ++l) {
    int nd = D_ * D_, nf = D_ * DFF_;
    transpose_bf16_kernel<<<cdiv(nd,256),256,0,stream>>>(Wq + (size_t)l*nd, WqT + (size_t)l*nd, D_, D_);
    transpose_bf16_kernel<<<cdiv(nd,256),256,0,stream>>>(Wk + (size_t)l*nd, WkT + (size_t)l*nd, D_, D_);
    transpose_bf16_kernel<<<cdiv(nd,256),256,0,stream>>>(Wv + (size_t)l*nd, WvT + (size_t)l*nd, D_, D_);
    transpose_bf16_kernel<<<cdiv(nd,256),256,0,stream>>>(Wo + (size_t)l*nd, WoT + (size_t)l*nd, D_, D_);
    transpose_bf16_kernel<<<cdiv(nf,256),256,0,stream>>>(W1 + (size_t)l*nf, W1T + (size_t)l*nf, D_, DFF_);
    transpose_bf16_kernel<<<cdiv(nf,256),256,0,stream>>>(W2 + (size_t)l*nf, W2T + (size_t)l*nf, DFF_, D_);
  }
  {
    int n = MS_ * DT_;
    cvt_bf16_kernel<<<cdiv(n,256),256,0,stream>>>(colv,  colbf,  n);
    cvt_bf16_kernel<<<cdiv(n,256),256,0,stream>>>(textv, textbf, n);
  }

  // ---- relation masks ----
  {
    int n = B_ * S_ * SW_;
    build_masks_kernel<<<cdiv(n,256),256,0,stream>>>(node, f2p, col, tab, pad, maskw);
  }

  dim3 gblk(32, 4);

  // ---- encoder ----
  {
    dim3 gg(D_/64, MS_/128);
    gemm_bf16_kernel<<<gg,gblk,0,stream>>>(colbf,  wcolT,  nullptr, tcol,  nullptr, MS_, D_, DT_, 5);
    gemm_bf16_kernel<<<gg,gblk,0,stream>>>(textbf, wtextT, nullptr, ttext, nullptr, MS_, D_, DT_, 5);
    encode_pointwise_kernel<<<MS_,256,0,stream>>>(tcol, ttext, b_col, w_num, b_num, b_text,
        w_dt, b_dt, w_bool, b_bool, ln_g, ln_b, memb, numv, dtv, boolv, sem, msk, pad, x);
  }

  // ---- transformer layers ----
  dim3 ggD(D_/64, MS_/128);
  dim3 ggF(DFF_/64, MS_/128);
  dim3 ablk(32, 4);
  dim3 agrid(S_/16/4, B_*H_);
  for (int l = 0; l < L_; ++l) {
    size_t od = (size_t)l * D_ * D_, of = (size_t)l * D_ * DFF_;
    ln_rows_kernel<<<MS_,256,0,stream>>>(x, aln_g + (size_t)l*D_, aln_b + (size_t)l*D_, hbuf, nullptr);
    gemm_bf16_kernel<<<ggD,gblk,0,stream>>>(hbuf, WqT + od, nullptr, nullptr, Qb,  MS_, D_, D_, 3);
    gemm_bf16_kernel<<<ggD,gblk,0,stream>>>(hbuf, WkT + od, nullptr, nullptr, Kb,  MS_, D_, D_, 3);
    gemm_bf16_kernel<<<ggD,gblk,0,stream>>>(hbuf, WvT + od, nullptr, nullptr, Vtb, MS_, D_, D_, 4);
    attention_kernel<<<agrid,ablk,0,stream>>>(Qb, Kb, Vtb, maskw, attnb);
    gemm_bf16_kernel<<<ggD,gblk,0,stream>>>(attnb, WoT + od, nullptr, x, nullptr, MS_, D_, D_, 0);
    ln_rows_kernel<<<MS_,256,0,stream>>>(x, fln_g + (size_t)l*D_, fln_b + (size_t)l*D_, hbuf, nullptr);
    gemm_bf16_kernel<<<ggF,gblk,0,stream>>>(hbuf, W1T + of, b1 + (size_t)l*DFF_, nullptr, ffnb, MS_, DFF_, D_, 2);
    gemm_bf16_kernel<<<ggD,gblk,0,stream>>>(ffnb, W2T + of, b2 + (size_t)l*D_, x, nullptr, MS_, D_, DFF_, 7);
  }

  // ---- final LN -> f32 output ----
  ln_rows_kernel<<<MS_,256,0,stream>>>(x, out_g, out_b, nullptr, (float*)d_out);
}
